// MatchLayer_56049323213398
// MI455X (gfx1250) — matrix operand order repaired and checked by reading, not compiled, not run on hardware
//
#include <hip/hip_runtime.h>

typedef __attribute__((ext_vector_type(16))) _Float16 v16h;
typedef __attribute__((ext_vector_type(8)))  _Float16 v8h_t;
typedef __attribute__((ext_vector_type(8)))  float    v8f;

namespace {
constexpr int BN   = 4;
constexpr int CINC = 3;
constexpr int HH   = 384;
constexpr int WW   = 768;
constexpr int CO   = 32;
constexpr int DMAX = 192;
constexpr int H2   = HH / 2;   // 192
constexpr int W2   = WW / 2;   // 384
constexpr int WT   = W2 / 16;  // 24 pixel-tiles per row
constexpr long long PE = (long long)BN * CO * H2 * W2; // 9,437,184 elems per 32-ch tensor

constexpr int ACOLS = 132;             // 130 used + pad
constexpr int ASLAB = 3 * ACOLS * 32;  // A slab halfs (12672)
constexpr int WHALF = 9 * 1024;        // packed 32->32 weights halfs (9216)
constexpr int LDSBYTES = (ASLAB + WHALF) * 2; // 43776 B dynamic LDS
}

// ---------------------------------------------------------------------------
// conv1: 3x3 stride-2 (3->32) + bias + ReLU, and its 1x1 stride-2 shortcut.
// Scalar VALU (27 MACs/out, <1% of total FLOPs). NCHW f32 in -> NHWC f16 out.
// ---------------------------------------------------------------------------
__global__ void conv1_s2_relu_kernel(const float* __restrict__ x,
                                     const float* __restrict__ w1,
                                     const float* __restrict__ b1,
                                     const float* __restrict__ wsc,
                                     const float* __restrict__ bsc,
                                     _Float16* __restrict__ t1,
                                     _Float16* __restrict__ sc)
{
    long long idx = (long long)blockIdx.x * blockDim.x + threadIdx.x;
    if (idx >= PE) return;
    int co = (int)(idx & 31);
    long long pix = idx >> 5;
    int w = (int)(pix % W2);
    int h = (int)((pix / W2) % H2);
    int b = (int)(pix / ((long long)W2 * H2));

    float acc = b1[co];
    #pragma unroll
    for (int ci = 0; ci < CINC; ++ci) {
        #pragma unroll
        for (int ky = 0; ky < 3; ++ky) {
            int hy = 2 * h + ky - 1;
            if (hy < 0 || hy >= HH) continue;
            #pragma unroll
            for (int kx = 0; kx < 3; ++kx) {
                int wx = 2 * w + kx - 1;
                if (wx < 0 || wx >= WW) continue;
                acc += x[((long long)(b * CINC + ci) * HH + hy) * WW + wx] *
                       w1[((co * CINC + ci) * 3 + ky) * 3 + kx];
            }
        }
    }
    t1[idx] = (_Float16)fmaxf(acc, 0.f);

    float acs = bsc[co];
    #pragma unroll
    for (int ci = 0; ci < CINC; ++ci)
        acs += x[((long long)(b * CINC + ci) * HH + 2 * h) * WW + 2 * w] * wsc[co * CINC + ci];
    sc[idx] = (_Float16)acs;
}

// ---------------------------------------------------------------------------
// Pack OIHW f32 weights into WMMA B-fragment order:
//   wpack[tap][chunk][nhalf][lane][elem], elem e of lane -> K = 8*(lane>>4)+(e&7)+16*(e>>3),
//   column N = nhalf*16 + (lane&15).   (B 32x16 f16 lane layout, wave32)
// ---------------------------------------------------------------------------
__global__ void pack_weights_kernel(const float* __restrict__ wsrc,
                                    _Float16* __restrict__ wpack,
                                    int K, int ntaps)
{
    int nchunks = K >> 5;
    int total = ntaps * nchunks * 1024;
    int idx = blockIdx.x * blockDim.x + threadIdx.x;
    if (idx >= total) return;
    int tap   = idx / (nchunks * 1024);
    int rem   = idx % (nchunks * 1024);
    int chunk = rem >> 10;
    int r2    = rem & 1023;
    int nh    = r2 >> 9;
    int lane  = (r2 >> 4) & 31;
    int e     = r2 & 15;
    int n = nh * 16 + (lane & 15);
    int k = chunk * 32 + ((lane >> 4) << 3) + (e & 7) + ((e >> 3) << 4);
    wpack[idx] = (_Float16)wsrc[((long long)n * K + k) * ntaps + tap];
}

// ---------------------------------------------------------------------------
// 32->32 3x3 conv, fully LDS-staged via async global->LDS copies.
// Block = 8 waves = 128 contiguous pixels of one row (grid 2304 = 4*192*3).
// Stage: 3 input rows x 130 cols x 32ch (zero-padded halo) + 18KB packed
// weights -> LDS with GLOBAL_LOAD_ASYNC_TO_LDS_B128 (ASYNCcnt), then a fully
// unrolled 9-tap loop of ds_load_b128 + v_wmma_f32_16x16x32_f16.
// ---------------------------------------------------------------------------
__global__ void __launch_bounds__(256)
conv32_wmma_lds_kernel(const _Float16* __restrict__ src,     // NHWC 32ch f16
                       const _Float16* __restrict__ wpack,   // [9][2][32][16]
                       const float* __restrict__ bias,
                       const _Float16* __restrict__ residual,// NHWC 32ch, nullable
                       _Float16* __restrict__ dst16,         // nullable
                       float* __restrict__ dst32,            // NCHW f32, nullable
                       int do_relu)
{
    extern __shared__ _Float16 smem[];  // [0,ASLAB): input slab, [ASLAB,+WHALF): weights
    const int tid = threadIdx.x;

    int bcol   = blockIdx.x % 3;
    int rowIdx = blockIdx.x / 3;
    int h = rowIdx % H2;
    int b = rowIdx / H2;
    int w0blk = bcol * 128;

    // 1) zero the input slab (covers the halo padding)
    v8h_t z8 = {};
    for (int v = tid; v < ASLAB / 8; v += 256)
        *(v8h_t*)(smem + v * 8) = z8;
    __syncthreads();

    // 2) async-copy valid input pixels: rows h-1..h+1, cols w0blk-1..w0blk+128
    for (int idx = tid; idx < 390 * 4; idx += 256) {
        int p   = idx >> 2;    // pixel within slab (3*130)
        int q   = idx & 3;     // 16-byte quarter of the 64B pixel vector
        int dyi = p / 130;
        int cc  = p % 130;
        int hy  = h - 1 + dyi;
        int wx  = w0blk - 1 + cc;
        if (hy >= 0 && hy < H2 && wx >= 0 && wx < W2) {
            unsigned int ldsb = (unsigned int)(((dyi * ACOLS + cc) * 32 + q * 8) * 2);
            unsigned long long ga = (unsigned long long)(const void*)
                (src + ((long long)(b * H2 + hy) * W2 + wx) * 32 + q * 8);
            asm volatile("global_load_async_to_lds_b128 %0, %1, off"
                         :: "v"(ldsb), "v"(ga) : "memory");
        }
    }
    // 3) async-copy packed weights (18KB)
    for (int v = tid; v < WHALF / 8; v += 256) {
        unsigned int ldsb = (unsigned int)((ASLAB + v * 8) * 2);
        unsigned long long ga = (unsigned long long)(const void*)(wpack + v * 8);
        asm volatile("global_load_async_to_lds_b128 %0, %1, off"
                     :: "v"(ldsb), "v"(ga) : "memory");
    }
    asm volatile("s_wait_asynccnt 0x0" ::: "memory");
    __syncthreads();

    // 4) WMMA main loop (EXEC all-1s: no divergent exits in this kernel)
    int lane = tid & 31;
    int widx = tid >> 5;
    int m    = lane & 15;
    int half = lane >> 4;
    int w0   = w0blk + widx * 16;

    const _Float16* sA = smem;
    const _Float16* sW = smem + ASLAB;

    v8f c0 = {0.f, 0.f, 0.f, 0.f, 0.f, 0.f, 0.f, 0.f};
    v8f c1 = {0.f, 0.f, 0.f, 0.f, 0.f, 0.f, 0.f, 0.f};

    #pragma unroll
    for (int tap = 0; tap < 9; ++tap) {
        int dy = tap / 3;                  // slab row 0..2
        int dx = tap % 3;                  // 0..2
        int cc = widx * 16 + m + dx;       // slab col of (wpix + dx - 1)
        const _Float16* ap = sA + (dy * ACOLS + cc) * 32;
        // 16-bit A 16x32 lane layout: elems 0..7 -> K=8*half.., 8..15 -> K=16+8*half..
        v8h_t lo = *(const v8h_t*)(ap + 8 * half);
        v8h_t hi = *(const v8h_t*)(ap + 16 + 8 * half);
        v16h a;
        #pragma unroll
        for (int e = 0; e < 8; ++e) { a[e] = lo[e]; a[e + 8] = hi[e]; }

        const _Float16* wp = sW + tap * 1024;
        v16h bf0 = *(const v16h*)(wp + lane * 16);        // cout 0..15
        v16h bf1 = *(const v16h*)(wp + 512 + lane * 16);  // cout 16..31

        c0 = __builtin_amdgcn_wmma_f32_16x16x32_f16(false, a, false, bf0,
                                                    (short)0, c0, false, false);
        c1 = __builtin_amdgcn_wmma_f32_16x16x32_f16(false, a, false, bf1,
                                                    (short)0, c1, false, false);
    }

    // 5) epilogue: bias + residual + ReLU; C/D layout: lane -> N=lane&15, M=r+8*half
    int n = lane & 15;
    float bv0 = bias[n];
    float bv1 = bias[n + 16];
    #pragma unroll
    for (int r = 0; r < 8; ++r) {
        int mm  = r + 8 * half;
        int wp2 = w0 + mm;
        long long po = (long long)(b * H2 + h) * W2 + wp2;
        float o0 = c0[r] + bv0;
        float o1 = c1[r] + bv1;
        if (residual) {
            o0 += (float)residual[po * 32 + n];
            o1 += (float)residual[po * 32 + n + 16];
        }
        if (do_relu) { o0 = fmaxf(o0, 0.f); o1 = fmaxf(o1, 0.f); }
        if (dst16) {
            dst16[po * 32 + n]      = (_Float16)o0;
            dst16[po * 32 + n + 16] = (_Float16)o1;
        }
        if (dst32) {
            dst32[((long long)(b * CO + n) * H2 + h) * W2 + wp2]      = o0;
            dst32[((long long)(b * CO + n + 16) * H2 + h) * W2 + wp2] = o1;
        }
    }
}

// ---------------------------------------------------------------------------
// Generic implicit-GEMM conv for the 224-input convs (virtual concat of
// src0 = block2_l [32ch] and src1 = corr volume [192ch], both NHWC f16).
// B fragments come from global (weights are L2-resident); corr volume
// (113MB f16) is L2-resident on MI455X's 192MB L2.
// ---------------------------------------------------------------------------
template <int NCHUNKS>
__global__ void conv_wmma_kernel(const _Float16* __restrict__ src0,
                                 const _Float16* __restrict__ src1,
                                 const _Float16* __restrict__ wpack,
                                 const float* __restrict__ bias,
                                 const _Float16* __restrict__ residual,
                                 _Float16* __restrict__ dst16,
                                 float* __restrict__ dst32,
                                 int ntaps, int do_relu)
{
    const int tiles = BN * H2 * WT;
    int wave = blockIdx.x * (blockDim.x >> 5) + (threadIdx.x >> 5);
    if (wave >= tiles) return;          // wave-uniform: EXEC all-1s for WMMA

    int lane = threadIdx.x & 31;
    int m    = lane & 15;
    int half = lane >> 4;

    int w0 = (wave % WT) * 16;
    int h  = (wave / WT) % H2;
    int b  =  wave / (WT * H2);
    int wpix = w0 + m;

    const int C1 = (NCHUNKS - 1) * 32;

    v8f c0 = {0.f, 0.f, 0.f, 0.f, 0.f, 0.f, 0.f, 0.f};
    v8f c1 = {0.f, 0.f, 0.f, 0.f, 0.f, 0.f, 0.f, 0.f};

    for (int tap = 0; tap < ntaps; ++tap) {
        int dy = (ntaps == 9) ? (tap / 3 - 1) : 0;
        int dx = (ntaps == 9) ? (tap % 3 - 1) : 0;
        int hy = h + dy;
        int wx = wpix + dx;
        bool valid = (hy >= 0) & (hy < H2) & (wx >= 0) & (wx < W2);
        int hyc = valid ? hy : 0;
        int wxc = valid ? wx : 0;
        long long pixoff = (long long)(b * H2 + hyc) * W2 + wxc;

        #pragma unroll
        for (int ch = 0; ch < NCHUNKS; ++ch) {
            const _Float16* ap = (ch == 0) ? (src0 + pixoff * 32)
                                           : (src1 + pixoff * C1 + (ch - 1) * 32);
            v8h_t lo = *(const v8h_t*)(ap + 8 * half);
            v8h_t hi = *(const v8h_t*)(ap + 16 + 8 * half);
            v16h a;
            #pragma unroll
            for (int e = 0; e < 8; ++e) {
                a[e]     = valid ? lo[e] : (_Float16)0.0f;  // v_cndmask, no divergence
                a[e + 8] = valid ? hi[e] : (_Float16)0.0f;
            }

            const _Float16* wp = wpack + (long long)(tap * NCHUNKS + ch) * 1024;
            v16h bf0 = *(const v16h*)(wp + lane * 16);
            v16h bf1 = *(const v16h*)(wp + 512 + lane * 16);

            c0 = __builtin_amdgcn_wmma_f32_16x16x32_f16(false, a, false, bf0,
                                                        (short)0, c0, false, false);
            c1 = __builtin_amdgcn_wmma_f32_16x16x32_f16(false, a, false, bf1,
                                                        (short)0, c1, false, false);
        }
    }

    int n = lane & 15;
    float bv0 = bias[n];
    float bv1 = bias[n + 16];
    #pragma unroll
    for (int r = 0; r < 8; ++r) {
        int mm  = r + 8 * half;
        int wp2 = w0 + mm;
        long long po = (long long)(b * H2 + h) * W2 + wp2;
        float o0 = c0[r] + bv0;
        float o1 = c1[r] + bv1;
        if (residual) {
            o0 += (float)residual[po * 32 + n];
            o1 += (float)residual[po * 32 + n + 16];
        }
        if (do_relu) { o0 = fmaxf(o0, 0.f); o1 = fmaxf(o1, 0.f); }
        if (dst16) {
            dst16[po * 32 + n]      = (_Float16)o0;
            dst16[po * 32 + n + 16] = (_Float16)o1;
        }
        if (dst32) {
            dst32[((long long)(b * CO + n) * H2 + h) * W2 + wp2]      = o0;
            dst32[((long long)(b * CO + n + 16) * H2 + h) * W2 + wp2] = o1;
        }
    }
}

// ---------------------------------------------------------------------------
// Correlation volume + leaky ReLU(0.1): corr[b,h,w,d] = mean_c L*R(w-d).
// Left pixel vector staged in LDS, one thread per disparity. Output NHWC f16
// with D=192 as the channel dim -> feeds conv_wmma<7> chunks 1..6 directly.
// ---------------------------------------------------------------------------
__global__ void corr_leaky_kernel(const _Float16* __restrict__ L,
                                  const _Float16* __restrict__ R,
                                  _Float16* __restrict__ corr)
{
    __shared__ float lsh[CO];
    long long pix = blockIdx.x;                 // pix = (b*H2 + h)*W2 + w
    int w = (int)(pix % W2);
    if (threadIdx.x < CO) lsh[threadIdx.x] = (float)L[pix * 32 + threadIdx.x];
    __syncthreads();

    int d = threadIdx.x;
    float acc = 0.f;
    if (w >= d) {                               // same row since w-d >= 0
        const _Float16* rp = R + (pix - d) * 32;
        #pragma unroll
        for (int c = 0; c < CO; ++c) acc += lsh[c] * (float)rp[c];
    }
    acc *= (1.0f / 32.0f);
    acc = (acc > 0.f) ? acc : 0.1f * acc;
    corr[pix * DMAX + d] = (_Float16)acc;
}

// ---------------------------------------------------------------------------
extern "C" void kernel_launch(void* const* d_in, const int* in_sizes, int n_in,
                              void* d_out, int out_size, void* d_ws, size_t ws_size,
                              hipStream_t stream)
{
    (void)in_sizes; (void)n_in; (void)out_size; (void)ws_size;

    const float* left  = (const float*)d_in[0];
    const float* right = (const float*)d_in[1];
    const float* b1_w1 = (const float*)d_in[2];
    const float* b1_b1 = (const float*)d_in[3];
    const float* b1_w2 = (const float*)d_in[4];
    const float* b1_b2 = (const float*)d_in[5];
    const float* b1_ws = (const float*)d_in[6];
    const float* b1_bs = (const float*)d_in[7];
    const float* b2_w1 = (const float*)d_in[8];
    const float* b2_b1 = (const float*)d_in[9];
    const float* b2_w2 = (const float*)d_in[10];
    const float* b2_b2 = (const float*)d_in[11];
    const float* cb_w1 = (const float*)d_in[12];
    const float* cb_b1 = (const float*)d_in[13];
    const float* cb_w2 = (const float*)d_in[14];
    const float* cb_b2 = (const float*)d_in[15];
    const float* cb_ws = (const float*)d_in[16];
    const float* cb_bs = (const float*)d_in[17];

    float* out = (float*)d_out;

    // ---- workspace layout (f16 elements) ----
    _Float16* ws0 = (_Float16*)d_ws;
    long long off = 0;
    _Float16* pk_b1w2 = ws0 + off; off += 9 * 1024;
    _Float16* pk_b2w1 = ws0 + off; off += 9 * 1024;
    _Float16* pk_b2w2 = ws0 + off; off += 9 * 1024;
    _Float16* pk_cbw1 = ws0 + off; off += 9 * 7 * 1024;
    _Float16* pk_cbw2 = ws0 + off; off += 9 * 1024;
    _Float16* pk_cbws = ws0 + off; off += 7 * 1024;
    _Float16* buf0  = ws0 + off; off += PE;       // scratch conv output
    _Float16* buf1  = ws0 + off; off += PE;       // shortcut
    _Float16* buf2  = ws0 + off; off += PE;       // block1
    _Float16* bl2l  = ws0 + off; off += PE;       // block2_l (f16 copy)
    _Float16* bl2r  = ws0 + off; off += PE;       // block2_r (f16 copy)
    _Float16* corrb = ws0 + off; off += 6 * PE;   // corr volume, NHWC 192ch

    auto pack = [&](const float* w, _Float16* dst, int K, int ntaps) {
        int total = ntaps * (K / 32) * 1024;
        pack_weights_kernel<<<(total + 255) / 256, 256, 0, stream>>>(w, dst, K, ntaps);
    };
    pack(b1_w2, pk_b1w2, 32, 9);
    pack(b2_w1, pk_b2w1, 32, 9);
    pack(b2_w2, pk_b2w2, 32, 9);
    pack(cb_w1, pk_cbw1, 224, 9);
    pack(cb_w2, pk_cbw2, 32, 9);
    pack(cb_ws, pk_cbws, 224, 1);

    const int conv1Blocks = (int)((PE + 255) / 256);   // 36864
    const int c32Blocks   = BN * H2 * 3;               // 2304: 8 waves x 16-px tiles
    const int wmmaBlocks  = (BN * H2 * WT) / 8;        // 2304 (generic kernel)
    const int corrBlocks  = BN * H2 * W2;              // one pixel per block

    // ---- left tower ----
    conv1_s2_relu_kernel<<<conv1Blocks, 256, 0, stream>>>(left, b1_w1, b1_b1, b1_ws, b1_bs, buf0, buf1);
    conv32_wmma_lds_kernel<<<c32Blocks, 256, LDSBYTES, stream>>>(buf0, pk_b1w2, b1_b2, buf1, buf2, nullptr, 1); // block1_l
    conv32_wmma_lds_kernel<<<c32Blocks, 256, LDSBYTES, stream>>>(buf2, pk_b2w1, b2_b1, nullptr, buf0, nullptr, 1);
    conv32_wmma_lds_kernel<<<c32Blocks, 256, LDSBYTES, stream>>>(buf0, pk_b2w2, b2_b2, buf2, bl2l, out, 1);      // block2_l

    // ---- right tower ----
    conv1_s2_relu_kernel<<<conv1Blocks, 256, 0, stream>>>(right, b1_w1, b1_b1, b1_ws, b1_bs, buf0, buf1);
    conv32_wmma_lds_kernel<<<c32Blocks, 256, LDSBYTES, stream>>>(buf0, pk_b1w2, b1_b2, buf1, buf2, nullptr, 1); // block1_r
    conv32_wmma_lds_kernel<<<c32Blocks, 256, LDSBYTES, stream>>>(buf2, pk_b2w1, b2_b1, nullptr, buf0, nullptr, 1);
    conv32_wmma_lds_kernel<<<c32Blocks, 256, LDSBYTES, stream>>>(buf0, pk_b2w2, b2_b2, buf2, bl2r, out + PE, 1); // block2_r

    // ---- correlation volume (+ leaky ReLU) ----
    corr_leaky_kernel<<<corrBlocks, DMAX, 0, stream>>>(bl2l, bl2r, corrb);

    // ---- corr ResBlock on virtual concat [bl2l | corrb] (224 ch) ----
    conv_wmma_kernel<7><<<wmmaBlocks, 256, 0, stream>>>(bl2l, corrb, pk_cbw1, cb_b1, nullptr, buf0, nullptr, 9, 1); // conv1+relu
    conv_wmma_kernel<7><<<wmmaBlocks, 256, 0, stream>>>(bl2l, corrb, pk_cbws, cb_bs, nullptr, buf1, nullptr, 1, 0); // 1x1 shortcut
    conv32_wmma_lds_kernel<<<c32Blocks, 256, LDSBYTES, stream>>>(buf0, pk_cbw2, cb_b2, buf1, nullptr, out + 2 * PE, 1);
}
